// GCN_41781441855659
// MI455X (gfx1250) — compile-verified, matrix operand order checked by reading, not run on hardware
//
#include <hip/hip_runtime.h>
#include <hip/hip_bf16.h>
#include <math.h>

// ---------------------------------------------------------------------------
// GATv2 GNN pipeline for MI455X (gfx1250, wave32).
// GEMMs use v_wmma_f32_16x16x32_f16 (f32 accum). Fast guard-free path for
// dims that are multiples of 16 (all but the final 250-col GEMM).
// Edge softmax: atomicMax on order-preserving uint keys, expf, atomicAdd.
// ---------------------------------------------------------------------------

typedef __attribute__((ext_vector_type(16))) _Float16 v16h;
typedef __attribute__((ext_vector_type(8)))  _Float16 v8h;
typedef __attribute__((ext_vector_type(8)))  float    v8f;

#define NEG_SLOPE 0.2f
#define EPS_BN 1e-5f

__device__ __forceinline__ unsigned f2key(float f) {
    unsigned u = __float_as_uint(f);
    return (u & 0x80000000u) ? ~u : (u | 0x80000000u);
}
__device__ __forceinline__ float key2f(unsigned k) {
    unsigned u = (k & 0x80000000u) ? (k & 0x7FFFFFFFu) : ~k;
    return __uint_as_float(u);
}

// ---------------- elementwise helpers ----------------
__global__ void k_f32_to_f16(const float* __restrict__ s, _Float16* __restrict__ d, int n) {
    for (int i = blockIdx.x * blockDim.x + threadIdx.x; i < n; i += gridDim.x * blockDim.x)
        d[i] = (_Float16)s[i];
}
__global__ void k_fill_u32(unsigned* __restrict__ p, unsigned v, int n) {
    for (int i = blockIdx.x * blockDim.x + threadIdx.x; i < n; i += gridDim.x * blockDim.x)
        p[i] = v;
}
__global__ void k_fill_f32(float* __restrict__ p, float v, int n) {
    for (int i = blockIdx.x * blockDim.x + threadIdx.x; i < n; i += gridDim.x * blockDim.x)
        p[i] = v;
}

// ---------------- WMMA GEMM (fast): C[M,Nout] = A[M,K]*W[Nout,K]^T + bias --
// Requires M%16==0 and Nout%16==0. 8 waves per block, one 16x16 tile per
// wave, no per-lane predication (EXEC stays all-ones around WMMA).
__global__ __launch_bounds__(256) void k_wmma_gemm_fast(
    const _Float16* __restrict__ A, const _Float16* __restrict__ W,
    const float* __restrict__ bias, float* __restrict__ C,
    int M, int K, int Nout) {
    const int wave = threadIdx.x >> 5;
    const int lane = threadIdx.x & 31;
    const int tm = blockIdx.x * 8 + wave;
    if (tm * 16 >= M) return;                  // wave-uniform exit
    const int tn = blockIdx.y;
    const int l15  = lane & 15;
    const int klow = (lane & 16) ? 8 : 0;      // lanes 16-31 hold K 8-15 / 24-31
    const _Float16* Ar = A + (size_t)(tm * 16 + l15) * K + klow;
    const _Float16* Wr = W + (size_t)(tn * 16 + l15) * K + klow;

    v8f acc = {};
    for (int k0 = 0; k0 < K; k0 += 32) {
        v16h a, b;
        ((v8h*)&a)[0] = *(const v8h*)(Ar + k0);
        ((v8h*)&a)[1] = *(const v8h*)(Ar + k0 + 16);
        ((v8h*)&b)[0] = *(const v8h*)(Wr + k0);
        ((v8h*)&b)[1] = *(const v8h*)(Wr + k0 + 16);
        acc = __builtin_amdgcn_wmma_f32_16x16x32_f16(
            false, a, false, b, (short)0, acc, false, false);
    }
    const int half = (lane >> 4) * 8;          // lanes 16-31 hold M = r+8
    const int col  = tn * 16 + l15;
    const float bv = bias[col];
    float* Crow = C + (size_t)(tm * 16 + half) * Nout + col;
    #pragma unroll
    for (int r = 0; r < 8; ++r)
        Crow[(size_t)r * Nout] = acc[r] + bv;
}

// ---------------- WMMA GEMM (col-guarded): for Nout not multiple of 16 -----
__global__ __launch_bounds__(32) void k_wmma_gemm_guard(
    const _Float16* __restrict__ A, const _Float16* __restrict__ W,
    const float* __restrict__ bias, float* __restrict__ C,
    int M, int K, int Nout) {
    const int tm = blockIdx.x, tn = blockIdx.y;
    const int lane = threadIdx.x;
    const int l15  = lane & 15;
    const int klow = (lane & 16) ? 8 : 0;
    const int row = tm * 16 + l15;
    const int col = tn * 16 + l15;
    const bool rok = row < M;
    const bool cok = col < Nout;

    v8f acc = {};
    for (int k0 = 0; k0 < K; k0 += 32) {
        v16h a, b;
        if (rok) {
            ((v8h*)&a)[0] = *(const v8h*)(A + (size_t)row * K + k0 + klow);
            ((v8h*)&a)[1] = *(const v8h*)(A + (size_t)row * K + k0 + klow + 16);
        } else {
            v8h z = {}; ((v8h*)&a)[0] = z; ((v8h*)&a)[1] = z;
        }
        if (cok) {
            ((v8h*)&b)[0] = *(const v8h*)(W + (size_t)col * K + k0 + klow);
            ((v8h*)&b)[1] = *(const v8h*)(W + (size_t)col * K + k0 + klow + 16);
        } else {
            v8h z = {}; ((v8h*)&b)[0] = z; ((v8h*)&b)[1] = z;
        }
        acc = __builtin_amdgcn_wmma_f32_16x16x32_f16(
            false, a, false, b, (short)0, acc, false, false);
    }
    const int half = (lane >> 4) * 8;
    const float bv = cok ? bias[col] : 0.f;
    for (int r = 0; r < 8; ++r) {
        int rr = tm * 16 + r + half;
        if (cok && rr < M)
            C[(size_t)rr * Nout + col] = acc[r] + bv;
    }
}

// ---------------- edge kernels ----------------
// logits[e] = att . leaky_relu(xl[src] + xr[dst] + edge_attr @ We^T)
__global__ void k_edge_logits(const float* __restrict__ xl, const float* __restrict__ xr,
                              const float* __restrict__ ea, const int* __restrict__ src,
                              const int* __restrict__ dst, const float* __restrict__ We,
                              const float* __restrict__ att, float* __restrict__ logits,
                              int E) {
    const int lane = threadIdx.x & 31;
    const int wid  = (blockIdx.x * blockDim.x + threadIdx.x) >> 5;
    const int nw   = (gridDim.x * blockDim.x) >> 5;
    const int f1 = lane, f2 = lane + 32;
    const float we10 = We[f1 * 2], we11 = We[f1 * 2 + 1], at1 = att[f1];
    const float we20 = We[f2 * 2], we21 = We[f2 * 2 + 1], at2 = att[f2];
    for (int e = wid; e < E; e += nw) {
        const int s = src[e], d = dst[e];
        const float a0 = ea[2 * e], a1 = ea[2 * e + 1];
        float m1 = xl[s * 64 + f1] + xr[d * 64 + f1] + a0 * we10 + a1 * we11;
        float m2 = xl[s * 64 + f2] + xr[d * 64 + f2] + a0 * we20 + a1 * we21;
        m1 = (m1 >= 0.f) ? m1 : NEG_SLOPE * m1;
        m2 = (m2 >= 0.f) ? m2 : NEG_SLOPE * m2;
        float acc = m1 * at1 + m2 * at2;
        #pragma unroll
        for (int off = 16; off > 0; off >>= 1) acc += __shfl_xor(acc, off, 32);
        if (lane == 0) logits[e] = acc;
    }
}

__global__ void k_seg_max(const float* __restrict__ logits, const int* __restrict__ dst,
                          unsigned* __restrict__ mxkey, int E) {
    for (int e = blockIdx.x * blockDim.x + threadIdx.x; e < E; e += gridDim.x * blockDim.x)
        atomicMax(&mxkey[dst[e]], f2key(logits[e]));
}

__global__ void k_edge_exp(const float* __restrict__ logits, const int* __restrict__ dst,
                           const unsigned* __restrict__ mxkey, float* __restrict__ exb,
                           float* __restrict__ denom, int E) {
    for (int e = blockIdx.x * blockDim.x + threadIdx.x; e < E; e += gridDim.x * blockDim.x) {
        const int d = dst[e];
        const float ex = expf(logits[e] - key2f(mxkey[d]));
        exb[e] = ex;
        atomicAdd(&denom[d], ex);
    }
}

__global__ void k_edge_alpha(float* __restrict__ exb, const int* __restrict__ dst,
                             const float* __restrict__ denom, int E) {
    for (int e = blockIdx.x * blockDim.x + threadIdx.x; e < E; e += gridDim.x * blockDim.x)
        exb[e] = exb[e] / denom[dst[e]];
}

__global__ void k_agg_init(float* __restrict__ agg, const float* __restrict__ cb, int n) {
    for (int i = blockIdx.x * blockDim.x + threadIdx.x; i < n; i += gridDim.x * blockDim.x)
        agg[i] = cb[i & 63];
}

// agg[dst] += alpha[e] * xl[src] ; one thread per (edge, feature)
__global__ void k_edge_agg(const float* __restrict__ xl, const float* __restrict__ alpha,
                           const int* __restrict__ src, const int* __restrict__ dst,
                           float* __restrict__ agg, int E) {
    const int n = E * 64;
    for (int i = blockIdx.x * blockDim.x + threadIdx.x; i < n; i += gridDim.x * blockDim.x) {
        const int e = i >> 6, f = i & 63;
        atomicAdd(&agg[dst[e] * 64 + f], alpha[e] * xl[src[e] * 64 + f]);
    }
}

// ---------------- BatchNorm ----------------
// stats[0..F) = sum, stats[F..2F) = sum of squares. F divides 256.
__global__ void k_bn_stats(const float* __restrict__ X, float* __restrict__ stats,
                           int M, int F) {
    const int t = threadIdx.x;
    const int f = t % F;
    const int rl = t / F;
    const int rpb = 256 / F;
    float s = 0.f, s2 = 0.f;
    for (int r = blockIdx.x * rpb + rl; r < M; r += gridDim.x * rpb) {
        const float v = X[(size_t)r * F + f];
        s += v; s2 += v * v;
    }
    atomicAdd(&stats[f], s);
    atomicAdd(&stats[F + f], s2);
}

__global__ void k_bn_finalize(const float* __restrict__ stats, const float* __restrict__ g,
                              const float* __restrict__ beta, float* __restrict__ ss,
                              int M, int F) {
    const int f = blockIdx.x * blockDim.x + threadIdx.x;
    if (f >= F) return;
    const float mean = stats[f] / (float)M;
    const float var  = stats[F + f] / (float)M - mean * mean;
    const float sc   = g[f] * rsqrtf(var + EPS_BN);
    ss[f]     = sc;
    ss[F + f] = beta[f] - mean * sc;
}

// out = prelu( (res? res : 0) + x*scale + shift , a )
__global__ void k_bn_apply_prelu(const float* __restrict__ X, const float* __restrict__ ss,
                                 const float* __restrict__ res, const float* __restrict__ aptr,
                                 float* __restrict__ out, int n, int F) {
    const float a = *aptr;
    for (int i = blockIdx.x * blockDim.x + threadIdx.x; i < n; i += gridDim.x * blockDim.x) {
        const int f = i % F;
        float y = X[i] * ss[f] + ss[F + f];
        if (res) y += res[i];
        out[i] = (y >= 0.f) ? y : a * y;
    }
}

// ---------------- pooling ----------------
__global__ void k_pool_max(const float* __restrict__ h, const int* __restrict__ batch,
                           unsigned* __restrict__ pkey, int N) {
    const int n = N * 64;
    for (int i = blockIdx.x * blockDim.x + threadIdx.x; i < n; i += gridDim.x * blockDim.x) {
        const int r = i >> 6, f = i & 63;
        atomicMax(&pkey[batch[r] * 64 + f], f2key(h[i]));
    }
}
__global__ void k_key_decode(const unsigned* __restrict__ pkey, float* __restrict__ p, int n) {
    for (int i = blockIdx.x * blockDim.x + threadIdx.x; i < n; i += gridDim.x * blockDim.x)
        p[i] = key2f(pkey[i]);
}

// ===========================================================================
// host side
// ===========================================================================
static inline void h_gemm(const _Float16* A, const _Float16* W, const float* bias,
                          float* C, int M, int K, int Nout, hipStream_t st) {
    if ((M % 16 == 0) && (Nout % 16 == 0)) {
        const int mt = M / 16;
        dim3 g((mt + 7) / 8, Nout / 16);
        k_wmma_gemm_fast<<<g, 256, 0, st>>>(A, W, bias, C, M, K, Nout);
    } else {
        dim3 g((M + 15) / 16, (Nout + 15) / 16);
        k_wmma_gemm_guard<<<g, 32, 0, st>>>(A, W, bias, C, M, K, Nout);
    }
}
static inline void h_tof16(const float* s, _Float16* d, int n, hipStream_t st) {
    int b = (n + 255) / 256; if (b > 2048) b = 2048;
    k_f32_to_f16<<<b, 256, 0, st>>>(s, d, n);
}

extern "C" void kernel_launch(void* const* d_in, const int* in_sizes, int n_in,
                              void* d_out, int out_size, void* d_ws, size_t ws_size,
                              hipStream_t stream) {
    // ---- inputs (setup_inputs order) ----
    const float* x        = (const float*)d_in[0];
    const int*   eidx     = (const int*)d_in[1];
    const float* eattr    = (const float*)d_in[2];
    const int*   batch    = (const int*)d_in[3];
    const float* pre_W    = (const float*)d_in[4];
    const float* pre_b    = (const float*)d_in[5];
    const float* pre_g    = (const float*)d_in[6];
    const float* pre_beta = (const float*)d_in[7];
    const float* pre_a    = (const float*)d_in[8];
    const float* blk_Wl   = (const float*)d_in[9];
    const float* blk_bl   = (const float*)d_in[10];
    const float* blk_Wr   = (const float*)d_in[11];
    const float* blk_br   = (const float*)d_in[12];
    const float* blk_We   = (const float*)d_in[13];
    const float* blk_att  = (const float*)d_in[14];
    const float* blk_cb   = (const float*)d_in[15];
    const float* blk_bng  = (const float*)d_in[16];
    const float* blk_bnb  = (const float*)d_in[17];
    const float* blk_pa   = (const float*)d_in[18];
    const float* post_W1  = (const float*)d_in[19];
    const float* post_b1  = (const float*)d_in[20];
    const float* post_g   = (const float*)d_in[21];
    const float* post_beta= (const float*)d_in[22];
    const float* post_a   = (const float*)d_in[23];
    const float* post_W2  = (const float*)d_in[24];
    const float* post_b2  = (const float*)d_in[25];

    const int N = in_sizes[3];          // 50000
    const int E = in_sizes[1] / 2;      // 800000
    const int NB2 = in_sizes[18];       // B*2 = 6
    const int G = in_sizes[21];         // 256
    const int C = in_sizes[25];         // 250
    const int H = 64;
    const int* src = eidx;
    const int* dst = eidx + E;

    // ---- workspace carving (256B aligned) ----
    char* base = (char*)d_ws;
    size_t off = 0;
    auto carve = [&](size_t bytes) -> char* {
        char* p = base + off;
        off = (off + bytes + 255) & ~(size_t)255;
        return p;
    };
    float*     hA     = (float*)carve((size_t)N * H * 4);
    float*     hB     = (float*)carve((size_t)N * H * 4);
    float*     xl     = (float*)carve((size_t)N * H * 4);
    float*     xr     = (float*)carve((size_t)N * H * 4);
    float*     agg    = (float*)carve((size_t)N * H * 4);
    _Float16*  h16    = (_Float16*)carve((size_t)N * H * 2);
    _Float16*  w16a   = (_Float16*)carve((size_t)256 * 256 * 2);
    _Float16*  w16b   = (_Float16*)carve((size_t)256 * 256 * 2);
    float*     logits = (float*)carve((size_t)E * 4);
    float*     exb    = (float*)carve((size_t)E * 4);
    unsigned*  mxkey  = (unsigned*)carve((size_t)N * 4);
    float*     denom  = (float*)carve((size_t)N * 4);
    float*     stats  = (float*)carve(512 * 4);
    float*     ss     = (float*)carve(512 * 4);
    unsigned*  pkey   = (unsigned*)carve((size_t)G * H * 4);
    float*     pooled = (float*)carve((size_t)G * H * 4);
    _Float16*  p16    = (_Float16*)carve((size_t)G * H * 2);
    float*     z1     = (float*)carve((size_t)G * 256 * 4);
    _Float16*  z116   = (_Float16*)carve((size_t)G * 256 * 2);
    (void)ws_size; (void)n_in; (void)out_size;

    const int nNH = N * H;
    const int GRID_E  = 2048;   // elementwise / edge grid-stride blocks
    const int GRID_EW = 8192;   // per-edge-wave and (edge,feature) kernels
    const unsigned NEGINF_KEY = 0x007FFFFFu;  // == f2key(-inf): ~bits(0xFF800000)

    auto bn_pass = [&](const float* X, int M, int F, const float* g,
                       const float* beta, const float* res, const float* aptr,
                       float* out) {
        k_fill_f32<<<2, 256, 0, stream>>>(stats, 0.f, 2 * F);
        k_bn_stats<<<1024, 256, 0, stream>>>(X, stats, M, F);
        k_bn_finalize<<<(F + 63) / 64, 64, 0, stream>>>(stats, g, beta, ss, M, F);
        k_bn_apply_prelu<<<GRID_E, 256, 0, stream>>>(X, ss, res, aptr, out, M * F, F);
    };

    // ---------------- pre layer: h = prelu(bn(x @ pre_W^T + pre_b)) --------
    h_tof16(x, h16, nNH, stream);
    h_tof16(pre_W, w16a, H * H, stream);
    h_gemm(h16, w16a, pre_b, agg, N, H, H, stream);
    bn_pass(agg, N, H, pre_g, pre_beta, nullptr, pre_a, hA);

    // ---------------- B blocks x 2 convs ----------------
    for (int c = 0; c < NB2; ++c) {
        const int j = c & 1;                      // conv index within block
        const float* hin = (j == 0) ? hA : hB;

        // node transforms: xl = hin@Wl^T + bl ; xr = hin@Wr^T + br
        h_tof16(hin, h16, nNH, stream);
        h_tof16(blk_Wl + (size_t)c * H * H, w16a, H * H, stream);
        h_tof16(blk_Wr + (size_t)c * H * H, w16b, H * H, stream);
        h_gemm(h16, w16a, blk_bl + c * H, xl, N, H, H, stream);
        h_gemm(h16, w16b, blk_br + c * H, xr, N, H, H, stream);

        // attention logits + segment softmax over dst
        k_edge_logits<<<GRID_EW, 256, 0, stream>>>(xl, xr, eattr, src, dst,
                                                   blk_We + c * H * 2,
                                                   blk_att + c * H, logits, E);
        k_fill_u32<<<GRID_E, 256, 0, stream>>>(mxkey, NEGINF_KEY, N);
        k_seg_max<<<GRID_E, 256, 0, stream>>>(logits, dst, mxkey, E);
        k_fill_f32<<<GRID_E, 256, 0, stream>>>(denom, 0.f, N);
        k_edge_exp<<<GRID_E, 256, 0, stream>>>(logits, dst, mxkey, exb, denom, E);
        k_edge_alpha<<<GRID_E, 256, 0, stream>>>(exb, dst, denom, E);

        // weighted aggregation (+ conv bias)
        k_agg_init<<<GRID_E, 256, 0, stream>>>(agg, blk_cb + c * H, nNH);
        k_edge_agg<<<GRID_EW, 256, 0, stream>>>(xl, exb, src, dst, agg, E);

        // epilogue
        if (j == 0) {
            // h = prelu(bn(agg))  -> hB   (hA kept as residual)
            bn_pass(agg, N, H, blk_bng + c * H, blk_bnb + c * H,
                    nullptr, blk_pa + c, hB);
        } else {
            // h = prelu(res + bn(agg)) -> hA
            bn_pass(agg, N, H, blk_bng + c * H, blk_bnb + c * H,
                    hA, blk_pa + c, hA);
        }
    }

    // ---------------- pooling: segment_max over batch ----------------
    k_fill_u32<<<64, 256, 0, stream>>>(pkey, NEGINF_KEY, G * H);
    k_pool_max<<<GRID_E, 256, 0, stream>>>(hA, batch, pkey, N);
    k_key_decode<<<64, 256, 0, stream>>>(pkey, pooled, G * H);

    // ---------------- post: z = prelu(bn(pooled@W1^T+b1)) ; out = z@W2^T+b2
    h_tof16(pooled, p16, G * H, stream);
    h_tof16(post_W1, w16a, 256 * H, stream);
    h_gemm(p16, w16a, post_b1, z1, G, H, 256, stream);
    bn_pass(z1, G, 256, post_g, post_beta, nullptr, post_a, z1);
    h_tof16(z1, z116, G * 256, stream);
    h_tof16(post_W2, w16b, C * 256, stream);
    h_gemm(z116, w16b, post_b2, (float*)d_out, G, 256, C, stream);
}
